// EmbSeqEncoder_14362370638146
// MI455X (gfx1250) — compile-verified, matrix operand order checked by reading, not run on hardware
//
#include <hip/hip_runtime.h>
#include <hip/hip_bf16.h>

typedef __attribute__((ext_vector_type(16))) __bf16 v16bf;
typedef __attribute__((ext_vector_type(8)))  __bf16 v8bf;
typedef __attribute__((ext_vector_type(2)))  __bf16 v2bf;
typedef __attribute__((ext_vector_type(8)))  float  v8f;

#define LDB 36          // u32 per LDS row (72 bf16 = 64 data + 8 pad)
#define LDB_BYTES 144

__device__ __forceinline__ unsigned pack2_bf16(float a, float b) {
  v2bf p;
  p[0] = (__bf16)a;     // RTNE; lowers to v_cvt_pk_bf16_f32
  p[1] = (__bf16)b;
  return __builtin_bit_cast(unsigned, p);
}

__device__ __forceinline__ uint4 pack8_bf16(const float4 f0, const float4 f1) {
  uint4 p;
  p.x = pack2_bf16(f0.x, f0.y);
  p.y = pack2_bf16(f0.z, f0.w);
  p.z = pack2_bf16(f1.x, f1.y);
  p.w = pack2_bf16(f1.z, f1.w);
  return p;
}

// Build a 16-element bf16 fragment from two 16B LDS runs (K {e0..e0+7} and {e0+16..e0+23})
__device__ __forceinline__ v16bf ld_frag(const char* rowBase, int e0) {
  v8bf lo = *(const v8bf*)(rowBase + e0 * 2);
  v8bf hi = *(const v8bf*)(rowBase + (e0 + 16) * 2);
  return __builtin_shufflevector(lo, hi, 0,1,2,3,4,5,6,7,8,9,10,11,12,13,14,15);
}

// ---------------- Kernel 0: exclusive cumsum of frag_lengths ----------------
__global__ __launch_bounds__(1024)
void scan_kernel(const int* __restrict__ lens, int* __restrict__ offs, int cnt) {
  __shared__ int s[1024];
  int t = threadIdx.x;
  int v = (t < cnt) ? lens[t] : 0;
  s[t] = v;
  __syncthreads();
  #pragma unroll
  for (int st = 1; st < 1024; st <<= 1) {
    int add = (t >= st) ? s[t - st] : 0;
    __syncthreads();
    s[t] += add;
    __syncthreads();
  }
  if (t < cnt) offs[t] = s[t] - v;   // exclusive scan
}

// -------- Kernel 1: beg/end tokens + zero padding (rows 1..len skipped) -----
__global__ __launch_bounds__(256)
void init_out_kernel(const float* __restrict__ beg, const float* __restrict__ endp,
                     const int* __restrict__ lens, float* __restrict__ out,
                     int D, int Lpad) {
  int row  = blockIdx.x;            // frag * Lpad + j
  int frag = row / Lpad;
  int j    = row - frag * Lpad;
  int len  = lens[frag];
  if (j >= 1 && j <= len) return;   // GEMM kernel fills these rows
  float* o = out + row * D;         // < 2^31 elements total
  for (int c = threadIdx.x; c < D; c += blockDim.x) {
    float v = 0.0f;
    if (j == 0)            v = beg[c];
    else if (j == len + 1) v = endp[c];
    o[c] = v;
  }
}

// --------- Kernel 2: fused GEMM -> scattered fragment rows (bf16 WMMA) ------
// Block = 256 threads (8 waves). Block tile: 32 rows x 256 cols.
// Wave tile: 32x32 (2 A-frags x 2 B-frags, 4 f32 accumulators).
// Double-buffered LDS, software-pipelined global->reg->LDS staging.
__global__ __launch_bounds__(256)
void gemm_frag_kernel(const float* __restrict__ A,     // [total, K] sent_embs
                      const float* __restrict__ Wm,    // [D, K]
                      const float* __restrict__ bias,  // [D]
                      const int* __restrict__ lens,
                      const int* __restrict__ offs,
                      float* __restrict__ out,         // [cnt, Lpad, D]
                      int total, int K, int D, int Lpad, int MTILES) {
  const int fb   = blockIdx.x;
  const int frag = fb / MTILES;
  const int mt   = fb - frag * MTILES;
  const int len  = lens[frag];
  if (mt * 32 >= len) return;       // uniform early-out per block
  const int off   = offs[frag];
  const int nbase = blockIdx.y * 256;

  __shared__ __align__(16) unsigned Alds[2][32 * LDB];    //  2 x  4.6 KB
  __shared__ __align__(16) unsigned Blds[2][256 * LDB];   //  2 x 36.9 KB

  const int tid   = threadIdx.x;
  const int lane  = tid & 31;
  const int wave  = tid >> 5;
  const int lhalf = lane >> 4;      // 0: lanes 0-15, 1: lanes 16-31
  const int lrow  = lane & 15;

  // --- staging assignments ---
  const int aRow = tid >> 3;                 // 0..31 (8 threads per row)
  const int aK   = (tid & 7) * 8;            // 8 floats per thread
  long long aSrc = (long long)off + mt * 32 + aRow;
  if (aSrc > (long long)total - 1) aSrc = (long long)total - 1;   // clamp (masked later)
  const float* aPtr = A + aSrc * K + aK;

  const int bCol = tid;                      // 0..255 (one column per thread)
  const float* bPtr = Wm + (long long)(nbase + bCol) * K;

  v8f acc00 = {}, acc01 = {}, acc10 = {}, acc11 = {};

  const int aSlot = aRow * LDB + (aK >> 1);
  const int bSlot = bCol * LDB;
  const char* aRow0 = (const char*)&Alds[0][0] + lrow * LDB_BYTES;
  const char* bRow0 = (const char*)&Blds[0][0] + (wave * 32 + lrow) * LDB_BYTES;
  const int aBufStep = 32 * LDB * 4;         // bytes between A buffers
  const int bBufStep = 256 * LDB * 4;        // bytes between B buffers

  float4 sa0, sa1, sb[16];

  // ---- prologue: stage chunk 0 ----
  sa0 = *(const float4*)(aPtr);
  sa1 = *(const float4*)(aPtr + 4);
  #pragma unroll
  for (int q = 0; q < 16; ++q) sb[q] = *(const float4*)(bPtr + q * 4);
  *(uint4*)&Alds[0][aSlot] = pack8_bf16(sa0, sa1);
  #pragma unroll
  for (int q = 0; q < 8; ++q)
    *(uint4*)&Blds[0][bSlot + q * 4] = pack8_bf16(sb[2 * q], sb[2 * q + 1]);
  __syncthreads();

  int p = 0;
  for (int k0 = 64; k0 < K; k0 += 64) {
    // ---- issue global loads for next chunk (latency hidden by WMMAs below) ----
    sa0 = *(const float4*)(aPtr + k0);
    sa1 = *(const float4*)(aPtr + k0 + 4);
    #pragma unroll
    for (int q = 0; q < 16; ++q) sb[q] = *(const float4*)(bPtr + k0 + q * 4);
    __builtin_prefetch(aPtr + k0 + 64, 0, 0);   // speculative, dropped if OOB
    __builtin_prefetch(bPtr + k0 + 64, 0, 0);

    // ---- compute current chunk from buffer p ----
    {
      const char* aB = aRow0 + p * aBufStep;
      const char* bB = bRow0 + p * bBufStep;
      #pragma unroll
      for (int ks = 0; ks < 64; ks += 32) {
        // lanes 0-15 hold K {0..7, 16..23}; lanes 16-31 hold K {8..15, 24..31}
        const int e0 = ks + lhalf * 8;
        v16bf bf0 = ld_frag(bB, e0);
        v16bf bf1 = ld_frag(bB + 16 * LDB_BYTES, e0);
        v16bf af0 = ld_frag(aB, e0);
        v16bf af1 = ld_frag(aB + 16 * LDB_BYTES, e0);
        acc00 = __builtin_amdgcn_wmma_f32_16x16x32_bf16(false, af0, false, bf0,
                                                        (short)0, acc00, false, false);
        acc01 = __builtin_amdgcn_wmma_f32_16x16x32_bf16(false, af0, false, bf1,
                                                        (short)0, acc01, false, false);
        acc10 = __builtin_amdgcn_wmma_f32_16x16x32_bf16(false, af1, false, bf0,
                                                        (short)0, acc10, false, false);
        acc11 = __builtin_amdgcn_wmma_f32_16x16x32_bf16(false, af1, false, bf1,
                                                        (short)0, acc11, false, false);
      }
    }

    // ---- convert + store next chunk into buffer p^1 ----
    {
      const int np = p ^ 1;
      *(uint4*)&Alds[np][aSlot] = pack8_bf16(sa0, sa1);
      #pragma unroll
      for (int q = 0; q < 8; ++q)
        *(uint4*)&Blds[np][bSlot + q * 4] = pack8_bf16(sb[2 * q], sb[2 * q + 1]);
    }
    __syncthreads();
    p ^= 1;
  }

  // ---- epilogue compute: last chunk ----
  {
    const char* aB = aRow0 + p * aBufStep;
    const char* bB = bRow0 + p * bBufStep;
    #pragma unroll
    for (int ks = 0; ks < 64; ks += 32) {
      const int e0 = ks + lhalf * 8;
      v16bf bf0 = ld_frag(bB, e0);
      v16bf bf1 = ld_frag(bB + 16 * LDB_BYTES, e0);
      v16bf af0 = ld_frag(aB, e0);
      v16bf af1 = ld_frag(aB + 16 * LDB_BYTES, e0);
      acc00 = __builtin_amdgcn_wmma_f32_16x16x32_bf16(false, af0, false, bf0,
                                                      (short)0, acc00, false, false);
      acc01 = __builtin_amdgcn_wmma_f32_16x16x32_bf16(false, af0, false, bf1,
                                                      (short)0, acc01, false, false);
      acc10 = __builtin_amdgcn_wmma_f32_16x16x32_bf16(false, af1, false, bf0,
                                                      (short)0, acc10, false, false);
      acc11 = __builtin_amdgcn_wmma_f32_16x16x32_bf16(false, af1, false, bf1,
                                                      (short)0, acc11, false, false);
    }
  }

  // ---- epilogue: C/D layout -> out[frag, 1+row, col] + bias (32-bit indexing) ----
  const int gcol0 = nbase + wave * 32 + lrow;        // N = lane % 16
  const int gcol1 = gcol0 + 16;
  const float bv0 = bias[gcol0];
  const float bv1 = bias[gcol1];
  const int rowTop = mt * 32 + lhalf * 8;            // M = e + 8*(lane/16) (+16 for frag 1)
  float* o = out + frag * (Lpad * D) + D + rowTop * D;   // row j = 1 + rowTop
  #pragma unroll
  for (int e = 0; e < 8; ++e) {
    if (rowTop + e < len) {
      o[e * D + gcol0] = acc00[e] + bv0;
      o[e * D + gcol1] = acc01[e] + bv1;
    }
    if (rowTop + 16 + e < len) {
      o[(16 + e) * D + gcol0] = acc10[e] + bv0;
      o[(16 + e) * D + gcol1] = acc11[e] + bv1;
    }
  }
}

extern "C" void kernel_launch(void* const* d_in, const int* in_sizes, int n_in,
                              void* d_out, int out_size, void* d_ws, size_t ws_size,
                              hipStream_t stream) {
  const float* sent = (const float*)d_in[0];
  const float* Wm   = (const float*)d_in[1];
  const float* bias = (const float*)d_in[2];
  const float* beg  = (const float*)d_in[3];
  const float* endp = (const float*)d_in[4];
  const int*   lens = (const int*)d_in[5];

  const int cnt  = in_sizes[5];
  const int D    = in_sizes[2];                       // 768
  const int K    = in_sizes[1] / D;                   // 1024
  const long long total = (long long)in_sizes[0] / K; // 131072
  const int Lpad = out_size / (cnt * D);              // max_frag_len + 2 = 130
  const int maxlen = Lpad - 2;
  const int MTILES = (maxlen + 31) / 32;              // 4

  int*   offs = (int*)d_ws;
  float* out  = (float*)d_out;

  scan_kernel<<<1, 1024, 0, stream>>>(lens, offs, cnt);
  init_out_kernel<<<cnt * Lpad, 256, 0, stream>>>(beg, endp, lens, out, D, Lpad);

  dim3 grid(cnt * MTILES, D / 256);                   // 768 / 256 = 3
  gemm_frag_kernel<<<grid, 256, 0, stream>>>(sent, Wm, bias, lens, offs, out,
                                             (int)total, K, D, Lpad, MTILES);
}
